// MBNet_35940286333418
// MI455X (gfx1250) — matrix operand order repaired and checked by reading, not compiled, not run on hardware
//
#include <hip/hip_runtime.h>
#include <hip/hip_bf16.h>
#include <stdint.h>

#define N_NODES  20000
#define N_EDGES  320000
#define N_GRAPHS 256
#define F_IN     78
#define F_IN_P   96        // F_IN padded to multiple of 32
#define GAT_DIM  128
#define HEADS    10
#define PROJ     256
#define MIXFP_D  3239
#define MIXFP_P  3264      // padded to multiple of 32
#define BERT_D   768

typedef __attribute__((ext_vector_type(16))) _Float16 v16h;
typedef __attribute__((ext_vector_type(8)))  float    v8f;

typedef __attribute__((address_space(3))) void lds_void_t;
__device__ __forceinline__ unsigned lds_off(const void* p) {
  return (unsigned)(uintptr_t)(lds_void_t*)p;   // LDS byte offset
}

// ---------- monotone float<->uint mapping for float atomicMax ----------
__device__ __forceinline__ unsigned f2mono(float f) {
  unsigned b = __float_as_uint(f);
  return (b & 0x80000000u) ? ~b : (b | 0x80000000u);
}
__device__ __forceinline__ float mono2f(unsigned u) {
  unsigned b = (u & 0x80000000u) ? (u & 0x7fffffffu) : ~u;
  return __uint_as_float(b);
}
#define MONO_NEG_INF 0x007fffffu   // f2mono(-inf)

// ===================== generic WMMA GEMM (async-to-LDS pipelined) ============
// C[M,N] = act(A[M,Kp](f16) @ Bt[N,Kp](f16)^T + bias)
// A row-major [M,Kp]; Bt = B transposed, row-major [N,Kp].
// Kp % 32 == 0 (zero padded at conversion), M % 32 == 0, N % 128 == 0.
// Block tile 32x128, 8 waves; wave tile 16x32 -> 2x v_wmma_f32_16x16x32_f16.
// Next K-tile is DMA'd into the alternate LDS buffer with
// GLOBAL_LOAD_ASYNC_TO_LDS_B64/B128 (ASYNCcnt) while current tile computes.
// Last iteration peeled -> straight-line loop body [wait][barrier][issue][wmma].
template<bool RELU, bool HALF_OUT, bool HAS_BIAS>
__global__ __launch_bounds__(256)
void wmma_gemm_f16(const _Float16* __restrict__ A, const _Float16* __restrict__ Bt,
                   const float* __restrict__ bias, void* __restrict__ Cout,
                   int M, int N, int Kp) {
  __shared__ _Float16 As[2][32][40];    // 80B row stride (16B multiple)
  __shared__ _Float16 Bs[2][128][48];   // 96B row stride (16B multiple)
  const int t    = threadIdx.x;
  const int lane = t & 31;
  const int wave = t >> 5;
  const int wm = wave >> 2;             // 0..1 : M sub-tile (16 rows)
  const int wn = wave & 3;              // 0..3 : N sub-tile (32 cols)
  const int m0 = blockIdx.x * 32;
  const int n0 = blockIdx.y * 128;

  // staging coordinates: A tile 32x32 (b64/lane), B tile 128x32 (2x b128/lane)
  const int ar = t >> 3, ac = (t & 7) * 4;
  const int br = t >> 1, bc = (t & 1) * 16;
  const _Float16* ap = A  + (size_t)(m0 + ar) * Kp + ac;
  const _Float16* bp = Bt + (size_t)(n0 + br) * Kp + bc;
  const unsigned ldsA0 = lds_off(&As[0][ar][ac]), ldsA1 = lds_off(&As[1][ar][ac]);
  const unsigned ldsB0 = lds_off(&Bs[0][br][bc]), ldsB1 = lds_off(&Bs[1][br][bc]);

  auto issue = [&](unsigned la, unsigned lb, int k0) {
    const uint64_t ga = (uint64_t)(uintptr_t)(ap + k0);
    const uint64_t gb = (uint64_t)(uintptr_t)(bp + k0);
    asm volatile("global_load_async_to_lds_b64 %0, %1, off"
                 :: "v"(la), "v"(ga) : "memory");
    asm volatile("global_load_async_to_lds_b128 %0, %1, off"
                 :: "v"(lb), "v"(gb) : "memory");
    // INST_OFFSET applies to both LDS and global address -> second 16B chunk
    asm volatile("global_load_async_to_lds_b128 %0, %1, off offset:16"
                 :: "v"(lb), "v"(gb) : "memory");
  };

  v8f acc0 = {}, acc1 = {};
  const int mm = lane & 15, hh = lane >> 4;
  const int nn = lane & 15, kb = (lane >> 4) * 16;

  auto compute = [&](int pb) {
    v16h af, bf0, bf1;
#pragma unroll
    for (int i = 0; i < 8; ++i) {
      af[i]     = As[pb][wm * 16 + mm][hh * 8 + i];
      af[i + 8] = As[pb][wm * 16 + mm][16 + hh * 8 + i];
    }
#pragma unroll
    for (int i = 0; i < 8; ++i) {
      bf0[i]     = Bs[pb][wn * 32 + nn][hh * 8 + i];
      bf0[i + 8] = Bs[pb][wn * 32 + nn][16 + hh * 8 + i];
      bf1[i]     = Bs[pb][wn * 32 + 16 + nn][hh * 8 + i];
      bf1[i + 8] = Bs[pb][wn * 32 + 16 + nn][16 + hh * 8 + i];
    }
    acc0 = __builtin_amdgcn_wmma_f32_16x16x32_f16(false, af, false, bf0, (short)0, acc0, false, false);
    acc1 = __builtin_amdgcn_wmma_f32_16x16x32_f16(false, af, false, bf1, (short)0, acc1, false, false);
  };

  const int nk = Kp >> 5;
  issue(ldsA0, ldsB0, 0);               // prologue: tile 0 -> buffer 0
  unsigned ia = ldsA1, ib = ldsB1;      // next issue target (XOR-toggled)
  const unsigned tglA = ldsA0 ^ ldsA1, tglB = ldsB0 ^ ldsB1;
  int pb = 0;
  for (int ki = 0; ki < nk - 1; ++ki) {
    asm volatile("s_wait_asynccnt 0x0" ::: "memory");  // this wave's DMA done
    __syncthreads();                                   // all waves' DMA done
    issue(ia, ib, (ki + 1) << 5);                      // prefetch next tile
    ia ^= tglA; ib ^= tglB;
    compute(pb);
    pb ^= 1;
  }
  asm volatile("s_wait_asynccnt 0x0" ::: "memory");    // peeled last tile
  __syncthreads();
  compute(pb);

  // ---- epilogue: C/D layout (lane -> N, VGPR r -> M row) ----
  const int mb = (lane >> 4) * 8;
  const int gn0 = n0 + wn * 32 + nn;
  const int gn1 = gn0 + 16;
  const float bv0 = HAS_BIAS ? bias[gn0] : 0.0f;
  const float bv1 = HAS_BIAS ? bias[gn1] : 0.0f;
#pragma unroll
  for (int r = 0; r < 8; ++r) {
    const int gm = m0 + wm * 16 + mb + r;
    float v0 = acc0[r] + bv0;
    float v1 = acc1[r] + bv1;
    if (RELU) { v0 = fmaxf(v0, 0.0f); v1 = fmaxf(v1, 0.0f); }
    if (HALF_OUT) {
      ((_Float16*)Cout)[(size_t)gm * N + gn0] = (_Float16)v0;
      ((_Float16*)Cout)[(size_t)gm * N + gn1] = (_Float16)v1;
    } else {
      ((float*)Cout)[(size_t)gm * N + gn0] = v0;
      ((float*)Cout)[(size_t)gm * N + gn1] = v1;
    }
  }
}

// ===================== conversion / padding =====================
__global__ void cast_pad(const float* __restrict__ src, _Float16* __restrict__ dst,
                         int rows, int cols, int colsPad) {
  size_t i = (size_t)blockIdx.x * blockDim.x + threadIdx.x;
  size_t n = (size_t)rows * colsPad;
  size_t stride = (size_t)gridDim.x * blockDim.x;
  for (; i < n; i += stride) {
    int r = (int)(i / colsPad), c = (int)(i % colsPad);
    dst[i] = (c < cols) ? (_Float16)src[(size_t)r * cols + c] : (_Float16)0.0f;
  }
}
// weight W [K, N] f32 -> Wt [N, Kpad] f16 (transposed, K zero-padded)
__global__ void wt_prep(const float* __restrict__ W, _Float16* __restrict__ Wt,
                        int K, int N, int Kpad) {
  size_t i = (size_t)blockIdx.x * blockDim.x + threadIdx.x;
  size_t n = (size_t)N * Kpad;
  size_t stride = (size_t)gridDim.x * blockDim.x;
  for (; i < n; i += stride) {
    int nn = (int)(i / Kpad), k = (int)(i % Kpad);
    Wt[i] = (k < K) ? (_Float16)W[(size_t)k * N + nn] : (_Float16)0.0f;
  }
}
__global__ void fill_u32(unsigned* p, unsigned v, size_t n) {
  size_t i = (size_t)blockIdx.x * blockDim.x + threadIdx.x;
  size_t stride = (size_t)gridDim.x * blockDim.x;
  for (; i < n; i += stride) p[i] = v;
}
__global__ void fill_f32(float* p, float v, size_t n) {
  size_t i = (size_t)blockIdx.x * blockDim.x + threadIdx.x;
  size_t stride = (size_t)gridDim.x * blockDim.x;
  for (; i < n; i += stride) p[i] = v;
}

// ===================== GAT pieces =====================
__global__ __launch_bounds__(256)
void gat_logits(const _Float16* __restrict__ Hh, const float* __restrict__ aS,
                const float* __restrict__ aD, float* __restrict__ as_,
                float* __restrict__ ad_, int Nn, int Hn, int C) {
  int wid  = (int)((blockIdx.x * (size_t)blockDim.x + threadIdx.x) >> 5);
  int lane = threadIdx.x & 31;
  if (wid >= Nn * Hn) return;
  int h = wid % Hn;
  const _Float16* hp = Hh + (size_t)wid * C;
  float s1 = 0.f, s2 = 0.f;
  for (int c = lane; c < C; c += 32) {
    float v = (float)hp[c];
    s1 += v * aS[h * C + c];
    s2 += v * aD[h * C + c];
  }
  for (int off = 16; off; off >>= 1) {
    s1 += __shfl_down(s1, off, 32);
    s2 += __shfl_down(s2, off, 32);
  }
  if (lane == 0) { as_[wid] = s1; ad_[wid] = s2; }
}

__device__ __forceinline__ void edge_sd(int e, const int* __restrict__ ei, int& s, int& d) {
  if (e < N_EDGES) { s = ei[e]; d = ei[N_EDGES + e]; }
  else             { s = d = e - N_EDGES; }   // appended self-loops
}

__global__ void gat_edge_max(const float* __restrict__ as_, const float* __restrict__ ad_,
                             const int* __restrict__ ei, unsigned* __restrict__ amax,
                             int Nn, int Hn) {
  int i = blockIdx.x * blockDim.x + threadIdx.x;
  int total = (N_EDGES + Nn) * Hn;
  if (i >= total) return;
  int e = i / Hn, h = i - e * Hn;
  int s, d; edge_sd(e, ei, s, d);
  float x = as_[s * Hn + h] + ad_[d * Hn + h];
  x = x > 0.f ? x : 0.2f * x;                   // leaky_relu(0.2)
  atomicMax(&amax[d * Hn + h], f2mono(x));
}

__global__ void gat_edge_exp(const float* __restrict__ as_, const float* __restrict__ ad_,
                             const int* __restrict__ ei, const unsigned* __restrict__ amax,
                             float* __restrict__ ew, float* __restrict__ denom,
                             int Nn, int Hn) {
  int i = blockIdx.x * blockDim.x + threadIdx.x;
  int total = (N_EDGES + Nn) * Hn;
  if (i >= total) return;
  int e = i / Hn, h = i - e * Hn;
  int s, d; edge_sd(e, ei, s, d);
  float x = as_[s * Hn + h] + ad_[d * Hn + h];
  x = x > 0.f ? x : 0.2f * x;
  float wv = __expf(x - mono2f(amax[d * Hn + h]));
  ew[i] = wv;
  atomicAdd(&denom[d * Hn + h], wv);
}

// wave per (edge, head): gather 128-ch source row, scatter-add into dst
__global__ __launch_bounds__(256)
void gat_edge_agg(const _Float16* __restrict__ Hh, const int* __restrict__ ei,
                  const float* __restrict__ ew, const float* __restrict__ denom,
                  float* __restrict__ agg, int Nn, int Hn, int C) {
  int wid  = (int)((blockIdx.x * (size_t)blockDim.x + threadIdx.x) >> 5);
  int lane = threadIdx.x & 31;
  int total = (N_EDGES + Nn) * Hn;
  if (wid >= total) return;
  int e = wid / Hn, h = wid - e * Hn;
  int s, d; edge_sd(e, ei, s, d);
  float wv = ew[wid] / (denom[d * Hn + h] + 1e-16f);
  const _Float16* hp = Hh + ((size_t)s * Hn + h) * C;
  float* op = agg + ((size_t)d * Hn + h) * C;
  __builtin_prefetch(hp + lane * 4, 0, 0);     // global_prefetch_b8
  for (int c = lane; c < C; c += 32)
    atomicAdd(op + c, (float)hp[c] * wv);
}

__global__ void elu_bias_to_half(const float* __restrict__ agg, const float* __restrict__ bias,
                                 _Float16* __restrict__ out, size_t n, int cols) {
  size_t i = (size_t)blockIdx.x * blockDim.x + threadIdx.x;
  if (i >= n) return;
  float v = agg[i] + bias[i % cols];
  v = v > 0.f ? v : (__expf(v) - 1.f);          // ELU
  out[i] = (_Float16)v;
}
__global__ void elu_bias_to_float(const float* __restrict__ agg, const float* __restrict__ bias,
                                  float* __restrict__ out, size_t n, int cols) {
  size_t i = (size_t)blockIdx.x * blockDim.x + threadIdx.x;
  if (i >= n) return;
  float v = agg[i] + bias[i % cols];
  out[i] = v > 0.f ? v : (__expf(v) - 1.f);
}

__global__ void pool_max(const float* __restrict__ h2, const int* __restrict__ batch,
                         unsigned* __restrict__ poolu, int Nn, int C) {
  int i = blockIdx.x * blockDim.x + threadIdx.x;
  if (i >= Nn * C) return;
  int n = i / C, c = i - n * C;
  atomicMax(&poolu[batch[n] * C + c], f2mono(h2[i]));
}
__global__ void pool_to_half(const unsigned* __restrict__ poolu, _Float16* __restrict__ out, int n) {
  int i = blockIdx.x * blockDim.x + threadIdx.x;
  if (i < n) out[i] = (_Float16)mono2f(poolu[i]);
}

__global__ void cast_to_half_off(const float* __restrict__ src, _Float16* __restrict__ dst,
                                 int rows, int cols, int dstStride, int colOff) {
  int i = blockIdx.x * blockDim.x + threadIdx.x;
  if (i >= rows * cols) return;
  int r = i / cols, c = i - r * cols;
  dst[(size_t)r * dstStride + colOff + c] = (_Float16)src[i];
}

// LayerNorm over 256 cols, write f16 at column offset into concat buffer
__global__ __launch_bounds__(256)
void ln_row_to_half(const float* __restrict__ X, const float* __restrict__ g,
                    const float* __restrict__ b, _Float16* __restrict__ dst,
                    int dstStride, int colOff) {
  __shared__ float red[256];
  int row = blockIdx.x, t = threadIdx.x;
  float v = X[row * 256 + t];
  red[t] = v; __syncthreads();
  for (int s = 128; s; s >>= 1) { if (t < s) red[t] += red[t + s]; __syncthreads(); }
  float mu = red[0] * (1.f / 256.f); __syncthreads();
  float dv = v - mu;
  red[t] = dv * dv; __syncthreads();
  for (int s = 128; s; s >>= 1) { if (t < s) red[t] += red[t + s]; __syncthreads(); }
  float var = red[0] * (1.f / 256.f);
  float y = dv * rsqrtf(var + 1e-5f) * g[t] + b[t];
  dst[(size_t)row * dstStride + colOff + t] = (_Float16)y;
}

// out[row] = LN(combined_row) . Wt + bt
__global__ __launch_bounds__(256)
void final_head(const float* __restrict__ comb, const float* __restrict__ g,
                const float* __restrict__ b, const float* __restrict__ Wt,
                const float* __restrict__ bt, float* __restrict__ out) {
  __shared__ float red[256];
  int row = blockIdx.x, t = threadIdx.x;
  float v = comb[row * 256 + t];
  red[t] = v; __syncthreads();
  for (int s = 128; s; s >>= 1) { if (t < s) red[t] += red[t + s]; __syncthreads(); }
  float mu = red[0] * (1.f / 256.f); __syncthreads();
  float dv = v - mu;
  red[t] = dv * dv; __syncthreads();
  for (int s = 128; s; s >>= 1) { if (t < s) red[t] += red[t + s]; __syncthreads(); }
  float var = red[0] * (1.f / 256.f); __syncthreads();
  float y = dv * rsqrtf(var + 1e-5f) * g[t] + b[t];
  red[t] = y * Wt[t]; __syncthreads();
  for (int s = 128; s; s >>= 1) { if (t < s) red[t] += red[t + s]; __syncthreads(); }
  if (t == 0) out[row] = red[0] + bt[0];
}

// ===================== host orchestration =====================
extern "C" void kernel_launch(void* const* d_in, const int* in_sizes, int n_in,
                              void* d_out, int out_size, void* d_ws, size_t ws_size,
                              hipStream_t stream) {
  const float* x      = (const float*)d_in[0];
  const int*   ei     = (const int*)  d_in[1];   // [2, N_EDGES]: src then dst
  const int*   batch  = (const int*)  d_in[2];
  const float* mixfp  = (const float*)d_in[3];
  const float* bert   = (const float*)d_in[4];
  const float* W1  = (const float*)d_in[5];
  const float* aS1 = (const float*)d_in[6];
  const float* aD1 = (const float*)d_in[7];
  const float* b1  = (const float*)d_in[8];
  const float* W2  = (const float*)d_in[9];
  const float* aS2 = (const float*)d_in[10];
  const float* aD2 = (const float*)d_in[11];
  const float* b2  = (const float*)d_in[12];
  const float* Wg  = (const float*)d_in[13];
  const float* bg  = (const float*)d_in[14];
  const float* Wm  = (const float*)d_in[15];
  const float* bm  = (const float*)d_in[16];
  const float* gm  = (const float*)d_in[17];
  const float* hm  = (const float*)d_in[18];
  const float* Wq  = (const float*)d_in[19];
  const float* bq  = (const float*)d_in[20];
  const float* gq  = (const float*)d_in[21];
  const float* hq  = (const float*)d_in[22];
  const float* Wc  = (const float*)d_in[23];
  const float* gt  = (const float*)d_in[24];
  const float* ht  = (const float*)d_in[25];
  const float* Wt  = (const float*)d_in[26];
  const float* bt  = (const float*)d_in[27];
  float* out = (float*)d_out;                    // [0,256): out ; [256,65792): combined

  // ---- carve workspace ----
  char* wsp = (char*)d_ws;
  auto alloc = [&](size_t bytes) -> void* {
    char* r = (char*)(((uintptr_t)wsp + 255) & ~(uintptr_t)255);
    wsp = r + bytes;
    return (void*)r;
  };
  _Float16* xh    = (_Float16*)alloc((size_t)N_NODES * F_IN_P * 2);
  _Float16* W1t   = (_Float16*)alloc((size_t)(HEADS * GAT_DIM) * F_IN_P * 2);
  _Float16* W2t   = (_Float16*)alloc((size_t)GAT_DIM * (HEADS * GAT_DIM) * 2);
  _Float16* Wgt   = (_Float16*)alloc((size_t)PROJ * GAT_DIM * 2);
  _Float16* Wmt   = (_Float16*)alloc((size_t)PROJ * MIXFP_P * 2);
  _Float16* Wqt   = (_Float16*)alloc((size_t)PROJ * BERT_D * 2);
  _Float16* Wct   = (_Float16*)alloc((size_t)PROJ * (3 * PROJ) * 2);
  _Float16* mixh  = (_Float16*)alloc((size_t)N_GRAPHS * MIXFP_P * 2);
  _Float16* berth = (_Float16*)alloc((size_t)N_GRAPHS * BERT_D * 2);
  _Float16* H1h   = (_Float16*)alloc((size_t)N_NODES * HEADS * GAT_DIM * 2);
  float*    as1   = (float*)alloc((size_t)N_NODES * HEADS * 4);
  float*    ad1   = (float*)alloc((size_t)N_NODES * HEADS * 4);
  unsigned* amax1 = (unsigned*)alloc((size_t)N_NODES * HEADS * 4);
  float*    den1  = (float*)alloc((size_t)N_NODES * HEADS * 4);
  float*    ew1   = (float*)alloc((size_t)(N_EDGES + N_NODES) * HEADS * 4);
  float*    agg1  = (float*)alloc((size_t)N_NODES * HEADS * GAT_DIM * 4);
  _Float16* out1h = (_Float16*)alloc((size_t)N_NODES * HEADS * GAT_DIM * 2);
  _Float16* H2h   = (_Float16*)alloc((size_t)N_NODES * GAT_DIM * 2);
  float*    as2   = (float*)alloc((size_t)N_NODES * 4);
  float*    ad2   = (float*)alloc((size_t)N_NODES * 4);
  unsigned* amax2 = (unsigned*)alloc((size_t)N_NODES * 4);
  float*    den2  = (float*)alloc((size_t)N_NODES * 4);
  float*    ew2   = (float*)alloc((size_t)(N_EDGES + N_NODES) * 4);
  float*    agg2  = (float*)alloc((size_t)N_NODES * GAT_DIM * 4);
  float*    h2f   = (float*)alloc((size_t)N_NODES * GAT_DIM * 4);
  unsigned* poolu = (unsigned*)alloc((size_t)N_GRAPHS * GAT_DIM * 4);
  _Float16* poolh = (_Float16*)alloc((size_t)N_GRAPHS * GAT_DIM * 2);
  float*    tmpS  = (float*)alloc((size_t)N_GRAPHS * PROJ * 4);
  float*    tmpM  = (float*)alloc((size_t)N_GRAPHS * PROJ * 4);
  float*    tmpB  = (float*)alloc((size_t)N_GRAPHS * PROJ * 4);
  _Float16* allf  = (_Float16*)alloc((size_t)N_GRAPHS * 3 * PROJ * 2);

  auto grd = [](size_t n) -> unsigned {
    size_t b = (n + 255) / 256;
    return (unsigned)(b > 16384 ? 16384 : b);
  };

  // ---- 0. precision conversion: pad A-side, transpose+pad weights ----
  cast_pad<<<grd((size_t)N_NODES * F_IN_P), 256, 0, stream>>>(x, xh, N_NODES, F_IN, F_IN_P);
  cast_pad<<<grd((size_t)N_GRAPHS * MIXFP_P), 256, 0, stream>>>(mixfp, mixh, N_GRAPHS, MIXFP_D, MIXFP_P);
  cast_pad<<<grd((size_t)N_GRAPHS * BERT_D), 256, 0, stream>>>(bert, berth, N_GRAPHS, BERT_D, BERT_D);
  wt_prep<<<grd((size_t)(HEADS * GAT_DIM) * F_IN_P), 256, 0, stream>>>(W1, W1t, F_IN, HEADS * GAT_DIM, F_IN_P);
  wt_prep<<<grd((size_t)GAT_DIM * HEADS * GAT_DIM), 256, 0, stream>>>(W2, W2t, HEADS * GAT_DIM, GAT_DIM, HEADS * GAT_DIM);
  wt_prep<<<grd((size_t)PROJ * GAT_DIM), 256, 0, stream>>>(Wg, Wgt, GAT_DIM, PROJ, GAT_DIM);
  wt_prep<<<grd((size_t)PROJ * MIXFP_P), 256, 0, stream>>>(Wm, Wmt, MIXFP_D, PROJ, MIXFP_P);
  wt_prep<<<grd((size_t)PROJ * BERT_D), 256, 0, stream>>>(Wq, Wqt, BERT_D, PROJ, BERT_D);
  wt_prep<<<grd((size_t)PROJ * 3 * PROJ), 256, 0, stream>>>(Wc, Wct, 3 * PROJ, PROJ, 3 * PROJ);

  // ---- 1. GEMM1: H1 = x @ W1  (f16 out) ----
  wmma_gemm_f16<false, true, false><<<dim3(N_NODES / 32, (HEADS * GAT_DIM) / 128), 256, 0, stream>>>(
      xh, W1t, nullptr, H1h, N_NODES, HEADS * GAT_DIM, F_IN_P);

  // ---- 2. layer-1 attention ----
  {
    int waves = N_NODES * HEADS;
    gat_logits<<<(waves + 7) / 8, 256, 0, stream>>>(H1h, aS1, aD1, as1, ad1, N_NODES, HEADS, GAT_DIM);
    size_t nh = (size_t)N_NODES * HEADS;
    fill_u32<<<grd(nh), 256, 0, stream>>>(amax1, MONO_NEG_INF, nh);
    fill_f32<<<grd(nh), 256, 0, stream>>>(den1, 0.f, nh);
    int total = (N_EDGES + N_NODES) * HEADS;
    gat_edge_max<<<(total + 255) / 256, 256, 0, stream>>>(as1, ad1, ei, amax1, N_NODES, HEADS);
    gat_edge_exp<<<(total + 255) / 256, 256, 0, stream>>>(as1, ad1, ei, amax1, ew1, den1, N_NODES, HEADS);
    size_t na = (size_t)N_NODES * HEADS * GAT_DIM;
    fill_f32<<<grd(na), 256, 0, stream>>>(agg1, 0.f, na);
    gat_edge_agg<<<(total + 7) / 8, 256, 0, stream>>>(H1h, ei, ew1, den1, agg1, N_NODES, HEADS, GAT_DIM);
    elu_bias_to_half<<<(unsigned)((na + 255) / 256), 256, 0, stream>>>(agg1, b1, out1h, na, HEADS * GAT_DIM);
  }

  // ---- 3. GEMM2: H2 = h1 @ W2  (f16 out) ----
  wmma_gemm_f16<false, true, false><<<dim3(N_NODES / 32, GAT_DIM / 128), 256, 0, stream>>>(
      out1h, W2t, nullptr, H2h, N_NODES, GAT_DIM, HEADS * GAT_DIM);

  // ---- 4. layer-2 attention (1 head) ----
  {
    gat_logits<<<(N_NODES + 7) / 8, 256, 0, stream>>>(H2h, aS2, aD2, as2, ad2, N_NODES, 1, GAT_DIM);
    fill_u32<<<(N_NODES + 255) / 256, 256, 0, stream>>>(amax2, MONO_NEG_INF, (size_t)N_NODES);
    fill_f32<<<(N_NODES + 255) / 256, 256, 0, stream>>>(den2, 0.f, (size_t)N_NODES);
    int total = N_EDGES + N_NODES;
    gat_edge_max<<<(total + 255) / 256, 256, 0, stream>>>(as2, ad2, ei, amax2, N_NODES, 1);
    gat_edge_exp<<<(total + 255) / 256, 256, 0, stream>>>(as2, ad2, ei, amax2, ew2, den2, N_NODES, 1);
    size_t na = (size_t)N_NODES * GAT_DIM;
    fill_f32<<<grd(na), 256, 0, stream>>>(agg2, 0.f, na);
    gat_edge_agg<<<(total + 7) / 8, 256, 0, stream>>>(H2h, ei, ew2, den2, agg2, N_NODES, 1, GAT_DIM);
    elu_bias_to_float<<<(unsigned)((na + 255) / 256), 256, 0, stream>>>(agg2, b2, h2f, na, GAT_DIM);
  }

  // ---- 5. global max pool per graph ----
  {
    int np = N_GRAPHS * GAT_DIM;
    fill_u32<<<(np + 255) / 256, 256, 0, stream>>>(poolu, MONO_NEG_INF, (size_t)np);
    pool_max<<<(N_NODES * GAT_DIM + 255) / 256, 256, 0, stream>>>(h2f, batch, poolu, N_NODES, GAT_DIM);
    pool_to_half<<<(np + 255) / 256, 256, 0, stream>>>(poolu, poolh, np);
  }

  // ---- 6. dense head ----
  wmma_gemm_f16<true, false, true><<<dim3(N_GRAPHS / 32, PROJ / 128), 256, 0, stream>>>(
      poolh, Wgt, bg, tmpS, N_GRAPHS, PROJ, GAT_DIM);                 // structure
  wmma_gemm_f16<true, false, true><<<dim3(N_GRAPHS / 32, PROJ / 128), 256, 0, stream>>>(
      mixh, Wmt, bm, tmpM, N_GRAPHS, PROJ, MIXFP_P);                  // relu(mix)
  wmma_gemm_f16<true, false, true><<<dim3(N_GRAPHS / 32, PROJ / 128), 256, 0, stream>>>(
      berth, Wqt, bq, tmpB, N_GRAPHS, PROJ, BERT_D);                  // relu(bert)

  cast_to_half_off<<<(N_GRAPHS * PROJ + 255) / 256, 256, 0, stream>>>(
      tmpS, allf, N_GRAPHS, PROJ, 3 * PROJ, 0);
  ln_row_to_half<<<N_GRAPHS, 256, 0, stream>>>(tmpM, gm, hm, allf, 3 * PROJ, PROJ);
  ln_row_to_half<<<N_GRAPHS, 256, 0, stream>>>(tmpB, gq, hq, allf, 3 * PROJ, 2 * PROJ);

  // combined = allf @ Wc  -> d_out[256:]
  wmma_gemm_f16<false, false, false><<<dim3(N_GRAPHS / 32, PROJ / 128), 256, 0, stream>>>(
      allf, Wct, nullptr, out + N_GRAPHS, N_GRAPHS, PROJ, 3 * PROJ);

  // out = LN(combined) @ Wt + bt -> d_out[0:256]
  final_head<<<N_GRAPHS, 256, 0, stream>>>(out + N_GRAPHS, gt, ht, Wt, bt, out);
}
